// GraphAttentionLayer_49246095016058
// MI455X (gfx1250) — matrix operand order repaired and checked by reading, not compiled, not run on hardware
//
#include <hip/hip_runtime.h>
#include <hip/hip_bf16.h>
#include <stdint.h>

typedef __attribute__((ext_vector_type(16))) __bf16 v16bf;
typedef __attribute__((ext_vector_type(8)))  __bf16 v8bf;
typedef __attribute__((ext_vector_type(8)))  float  v8f;

#define GAT_B    8
#define GAT_N    2048
#define GAT_FIN  256
#define GAT_FOUT 128
#define GAT_ALPHA 0.01f
#define GAT_NEGINF (-1000000000000.0f)

// CDNA5 async global->LDS copy (ASYNCcnt tracked, no VGPR round-trip).
// lds_off: byte offset within workgroup LDS (flat addr low 32 bits).
__device__ __forceinline__ void async_load_b128(uint32_t lds_off, const void* gaddr) {
    asm volatile("global_load_async_to_lds_b128 %0, %1, off"
                 :: "v"(lds_off), "v"(gaddr) : "memory");
}
__device__ __forceinline__ void wait_async0() {
    asm volatile("s_wait_asynccnt 0x0" ::: "memory");
}
__device__ __forceinline__ uint32_t lds_addr(const void* p) {
    return (uint32_t)(uintptr_t)p;   // flat LDS address: addr[31:0] = LDS offset
}

// ---------------------------------------------------------------------------
// Kernel 0: W (256x128 f32, k-major) -> Wt (128x256 bf16, f-major)
// ---------------------------------------------------------------------------
__global__ void k_convW(const float* __restrict__ W, __bf16* __restrict__ Wt) {
    int idx = blockIdx.x * 256 + threadIdx.x;          // 32768 total
    if (idx < GAT_FIN * GAT_FOUT) {
        int f = idx / GAT_FIN;
        int k = idx % GAT_FIN;
        Wt[idx] = (__bf16)W[k * GAT_FOUT + f];
    }
}

// ---------------------------------------------------------------------------
// Kernel 1: h = inp @ W via v_wmma_f32_16x16x32_bf16.
// One block = one 16-row tile (8 waves = 8 F_out col-tiles).
// Emits ht (bf16, [B][FOUT][N] f-major) and s1,s2 (f32).
// ---------------------------------------------------------------------------
__global__ void __launch_bounds__(256) k_gemm1(
        const float*  __restrict__ inp,   // [16384][256] f32
        const __bf16* __restrict__ Wt,    // [128][256] bf16
        const float*  __restrict__ a,     // [256] f32
        __bf16*       __restrict__ ht,    // [8][128][2048] bf16
        float*        __restrict__ s1g,   // [16384]
        float*        __restrict__ s2g) { // [16384]
    __shared__ __align__(32) __bf16 sA[16][48];    // 16 rows x 32 K (padded)
    __shared__ __align__(32) __bf16 sB[128][48];   // 128 f  x 32 K (padded)
    __shared__ float s1t[16], s2t[16];

    const int tid  = threadIdx.x;
    const int wid  = tid >> 5;
    const int lane = tid & 31;
    const int m    = lane & 15;
    const int hi   = lane >> 4;
    const int i0   = blockIdx.x * 16;              // global row base
    const int b    = blockIdx.x >> 7;
    const int n0   = (blockIdx.x & 127) * 16;      // row base within batch

    if (tid < 16) { s1t[tid] = 0.f; s2t[tid] = 0.f; }

    v8f c = {};
    for (int ks = 0; ks < 8; ++ks) {               // K = 256 in steps of 32
        __syncthreads();
        {   // stage A tile: 16x32 f32 -> bf16 (needs VALU convert; manual path)
            int idx = tid * 2;
            int rr  = idx >> 5;
            int cc  = idx & 31;
            const float2 v = *(const float2*)(inp + (size_t)(i0 + rr) * GAT_FIN + ks * 32 + cc);
            sA[rr][cc]     = (__bf16)v.x;
            sA[rr][cc + 1] = (__bf16)v.y;
        }
        {   // stage B tile: 128x32 bf16, pure copy -> async global->LDS
            int f = tid >> 1, half = tid & 1;
            const char* g = (const char*)(Wt + (size_t)f * GAT_FIN + ks * 32 + half * 16);
            uint32_t l = lds_addr(&sB[f][half * 16]);
            async_load_b128(l,      g);
            async_load_b128(l + 16, g + 16);
        }
        wait_async0();
        __syncthreads();
        // A fragment: lane m holds row m, K = 8*hi+0..7 then 16+8*hi+0..7
        v16bf Av;
        const v8bf a0 = *(const v8bf*)&sA[m][8 * hi];
        const v8bf a1 = *(const v8bf*)&sA[m][16 + 8 * hi];
        for (int e = 0; e < 8; ++e) { Av[e] = a0[e]; Av[e + 8] = a1[e]; }
        // B fragment: lane holds col f = 16*wid+m, K = 8*hi+0..7 then 16+8*hi+0..7
        v16bf Bv;
        const v8bf b0 = *(const v8bf*)&sB[16 * wid + m][8 * hi];
        const v8bf b1 = *(const v8bf*)&sB[16 * wid + m][16 + 8 * hi];
        for (int e = 0; e < 8; ++e) { Bv[e] = b0[e]; Bv[e + 8] = b1[e]; }
        c = __builtin_amdgcn_wmma_f32_16x16x32_bf16(false, Av, false, Bv,
                                                    (short)0, c, false, false);
    }

    // store h^T tile: c[r] = h[row n0 + 8*hi + r][f], rows contiguous -> 1x b128
    const int f = 16 * wid + m;
    union { __bf16 h[8]; uint4 u; } pk;
    for (int r = 0; r < 8; ++r) pk.h[r] = (__bf16)c[r];
    *(uint4*)&ht[((size_t)(b * GAT_FOUT + f)) * GAT_N + n0 + 8 * hi] = pk.u;

    // s1/s2 partial reduction (f32 precision, LDS atomics across f)
    const float a1f = a[f];
    const float a2f = a[GAT_FOUT + f];
    for (int r = 0; r < 8; ++r) {
        atomicAdd(&s1t[r + 8 * hi], c[r] * a1f);
        atomicAdd(&s2t[r + 8 * hi], c[r] * a2f);
    }
    __syncthreads();
    if (tid < 16)      s1g[i0 + tid] = s1t[tid];
    else if (tid < 32) s2g[i0 + (tid - 16)] = s2t[tid - 16];
}

// ---------------------------------------------------------------------------
// Kernel 2: fused masked-leaky-softmax @ h (flash style), ReLU, store.
// Block = 8 waves covering 128 rows of one batch; h^T chunks double-buffered
// in LDS via async global->LDS loads overlapping score math + WMMA.
// ---------------------------------------------------------------------------
__global__ void __launch_bounds__(256) k_flash(
        const int*    __restrict__ adj,   // [8][2048][2048]
        const __bf16* __restrict__ ht,    // [8][128][2048]
        const float*  __restrict__ s1g,
        const float*  __restrict__ s2g,
        float*        __restrict__ out) { // [8][2048][128]
    __shared__ __align__(32) __bf16 sH[2][128][48];   // double-buffered h^T chunk

    const int tid  = threadIdx.x;
    const int wid  = tid >> 5;
    const int lane = tid & 31;
    const int m    = lane & 15;
    const int hi   = lane >> 4;
    const int rb   = blockIdx.x;                   // 0..15 row-block
    const int b    = blockIdx.y;                   // 0..7 batch
    const int nrow = rb * 128 + wid * 16 + m;      // this lane's P-row
    const int grow = b * GAT_N + nrow;

    const int*   adjrow = adj + (size_t)grow * GAT_N;
    const float* s2row  = s2g + (size_t)b * GAT_N;
    const float  s1v    = s1g[grow];

    // per-thread staging coordinates: 32B of one h^T row per thread
    const int fstage = tid >> 1, hstage = tid & 1;
    const char* gstage = (const char*)(ht + ((size_t)(b * GAT_FOUT + fstage)) * GAT_N + hstage * 16);

    // preload chunk 0 into buffer 0
    {
        uint32_t l = lds_addr(&sH[0][fstage][hstage * 16]);
        async_load_b128(l,      gstage);
        async_load_b128(l + 16, gstage + 16);
    }
    wait_async0();
    __syncthreads();

    float rowmax = -INFINITY;
    float rowsum = 0.f;
    v8f acc[8] = {};

    for (int jb = 0; jb < GAT_N; jb += 32) {
        const int cur = (jb >> 5) & 1;
        const int nxt = cur ^ 1;

        // kick off async stage of the next chunk; overlaps all math below
        if (jb + 32 < GAT_N) {
            uint32_t l = lds_addr(&sH[nxt][fstage][hstage * 16]);
            const char* g = gstage + (size_t)(jb + 32) * sizeof(__bf16);
            async_load_b128(l,      g);
            async_load_b128(l + 16, g + 16);
            __builtin_prefetch(adjrow + jb + 32, 0, 1);
        }

        // this lane's 16 columns match the 16-bit A-matrix VGPR layout:
        // K = 8*hi + 0..7 and K = 16 + 8*hi + 0..7
        const int c0 = jb + 8 * hi;
        const int c1 = jb + 16 + 8 * hi;
        int4 q0 = *(const int4*)(adjrow + c0);
        int4 q1 = *(const int4*)(adjrow + c0 + 4);
        int4 q2 = *(const int4*)(adjrow + c1);
        int4 q3 = *(const int4*)(adjrow + c1 + 4);
        float4 t0 = *(const float4*)(s2row + c0);
        float4 t1 = *(const float4*)(s2row + c0 + 4);
        float4 t2 = *(const float4*)(s2row + c1);
        float4 t3 = *(const float4*)(s2row + c1 + 4);

        const int   qa[16] = {q0.x,q0.y,q0.z,q0.w, q1.x,q1.y,q1.z,q1.w,
                              q2.x,q2.y,q2.z,q2.w, q3.x,q3.y,q3.z,q3.w};
        const float ta[16] = {t0.x,t0.y,t0.z,t0.w, t1.x,t1.y,t1.z,t1.w,
                              t2.x,t2.y,t2.z,t2.w, t3.x,t3.y,t3.z,t3.w};
        float z[16];
        for (int e = 0; e < 16; ++e) {
            float s = s1v + ta[e];
            s = (s > 0.f) ? s : GAT_ALPHA * s;
            z[e] = (qa[e] > 0) ? s : GAT_NEGINF;
        }

        // online softmax: lanes (m) and (m+16) cover disjoint halves of the row
        float tmax = z[0];
        for (int e = 1; e < 16; ++e) tmax = fmaxf(tmax, z[e]);
        tmax = fmaxf(tmax, __shfl_xor(tmax, 16, 32));
        const float newmax = fmaxf(rowmax, tmax);
        const float corr   = __expf(rowmax - newmax);
        rowmax = newmax;

        v16bf Av;
        float tsum = 0.f;
        for (int e = 0; e < 16; ++e) {
            float p = __expf(z[e] - newmax);
            tsum += p;
            Av[e] = (__bf16)p;
        }
        tsum += __shfl_xor(tsum, 16, 32);
        rowsum = rowsum * corr + tsum;

        // rescale f32 accumulator (row of acc VGPR r is r + 8*hi)
        for (int r = 0; r < 8; ++r) {
            float cf = __shfl(corr, r + 8 * hi, 32);
            for (int t = 0; t < 8; ++t) acc[t][r] *= cf;
        }

        // P (16x32 bf16) @ h-chunk (32x128 bf16) += acc (16x128 f32)
        for (int t = 0; t < 8; ++t) {
            v16bf Bv;
            const v8bf hb0 = *(const v8bf*)&sH[cur][16 * t + m][8 * hi];
            const v8bf hb1 = *(const v8bf*)&sH[cur][16 * t + m][16 + 8 * hi];
            for (int e = 0; e < 8; ++e) { Bv[e] = hb0[e]; Bv[e + 8] = hb1[e]; }
            acc[t] = __builtin_amdgcn_wmma_f32_16x16x32_bf16(false, Av, false, Bv,
                                                             (short)0, acc[t], false, false);
        }

        // next buffer must be fully landed before anyone reads it
        wait_async0();
        __syncthreads();
    }

    // normalize, ReLU, store
    for (int r = 0; r < 8; ++r) {
        float rs  = __shfl(rowsum, r + 8 * hi, 32);
        float inv = 1.f / rs;
        int n = rb * 128 + wid * 16 + r + 8 * hi;
        float* orow = out + ((size_t)(b * GAT_N + n)) * GAT_FOUT;
        for (int t = 0; t < 8; ++t) {
            float v = acc[t][r] * inv;
            orow[16 * t + m] = (v > 0.f) ? v : 0.f;
        }
    }
}

// ---------------------------------------------------------------------------
extern "C" void kernel_launch(void* const* d_in, const int* in_sizes, int n_in,
                              void* d_out, int out_size, void* d_ws, size_t ws_size,
                              hipStream_t stream) {
    (void)in_sizes; (void)n_in; (void)out_size; (void)ws_size;
    const float* inp = (const float*)d_in[0];
    const int*   adj = (const int*)d_in[1];
    const float* W   = (const float*)d_in[2];
    const float* a   = (const float*)d_in[3];
    float* out = (float*)d_out;

    char* ws = (char*)d_ws;
    __bf16* Wt = (__bf16*)(ws);                                         // 64 KB
    __bf16* ht = (__bf16*)(ws + 65536);                                 // 4 MB
    float*  s1 = (float*)(ws + 65536 + (size_t)GAT_B * GAT_FOUT * GAT_N * 2);
    float*  s2 = s1 + GAT_B * GAT_N;

    k_convW<<<128, 256, 0, stream>>>(W, Wt);
    k_gemm1<<<1024, 256, 0, stream>>>(inp, Wt, a, ht, s1, s2);
    k_flash<<<dim3(16, 8), 256, 0, stream>>>(adj, ht, s1, s2, out);
}
